// TransformerBlock_25864293057024
// MI455X (gfx1250) — matrix operand order repaired and checked by reading, not compiled, not run on hardware
//
#include <hip/hip_runtime.h>

#define DEV static __device__ __forceinline__

typedef __bf16 bf16;
typedef __attribute__((ext_vector_type(16))) __bf16 v16bf;
typedef __attribute__((ext_vector_type(8)))  float  v8f;

static constexpr int Dm   = 1024;  // embed dim
static constexpr int Hh   = 16;    // heads
static constexpr int DHd  = 64;    // head dim
static constexpr int DFFd = 4096;  // ffn dim
static constexpr int Ss   = 2048;  // seq len
static constexpr int NTOK = 4096;  // B*S

union Frag {
  v16bf v;
  uint4 q[2];
};

DEV bf16 f2bf(float f) {
  unsigned u = __builtin_bit_cast(unsigned, f);
  unsigned r = (u + 0x7FFFu + ((u >> 16) & 1u)) >> 16;  // round-to-nearest-even
  unsigned short h = (unsigned short)r;
  return __builtin_bit_cast(bf16, h);
}

DEV v8f zero8() {
  v8f z;
  #pragma unroll
  for (int i = 0; i < 8; ++i) z[i] = 0.f;
  return z;
}

DEV v8f wmma_bf16(const Frag& a, const Frag& b, v8f c) {
  // D = A(16x32 bf16) * B(32x16 bf16) + C(f32), f32 accumulate
  return __builtin_amdgcn_wmma_f32_16x16x32_bf16(
      false, a.v, false, b.v, (short)0, c, false, false);
}

// ---------------------------------------------------------------------------
// elementwise f32 -> bf16
__global__ __launch_bounds__(256) void cvt_bf16_kernel(const float* __restrict__ in,
                                                       bf16* __restrict__ out, int n) {
  int i = blockIdx.x * 256 + threadIdx.x;
  if (i < n) out[i] = f2bf(in[i]);
}

// transpose + convert: in f32 [Z][R][C] -> out bf16 [Z][C][R]
__global__ __launch_bounds__(256) void transpose_cvt_kernel(const float* __restrict__ in,
                                                            bf16* __restrict__ out,
                                                            int R, int C) {
  int z = blockIdx.y;
  long i = (long)blockIdx.x * 256 + threadIdx.x;
  long RC = (long)R * C;
  if (i >= RC) return;
  int r = (int)(i / C), c = (int)(i % C);
  out[(size_t)z * RC + (size_t)c * R + r] = f2bf(in[(size_t)z * RC + i]);
}

// ---------------------------------------------------------------------------
// Generic bf16 GEMM:  C[M,N] = A[M,K] * BT[N,K]^T + bias
//   block = 256 threads = 8 waves (4 M x 2 N); wave tile 32x64; block 128x128.
//   Software-pipelined: next K-step's 12 b128 loads issued before the current
//   8 WMMAs, so s_wait only drains the older load group (load/WMMA overlap).
//   mode 0: bf16 store   1: relu+bf16 store   2: f32 store
//   mode 3: bf16 store transposed per-head (V^T layout [B,H,DH,S])
struct Tile {
  Frag a[2];
  Frag b[4];
};

DEV void load_tile(Tile& t, const bf16* arow0, const bf16* arow1,
                   const bf16* b0, int K, int k) {
  t.a[0].q[0] = *(const uint4*)(arow0 + k);
  t.a[0].q[1] = *(const uint4*)(arow0 + k + 16);
  t.a[1].q[0] = *(const uint4*)(arow1 + k);
  t.a[1].q[1] = *(const uint4*)(arow1 + k + 16);
  #pragma unroll
  for (int nt = 0; nt < 4; ++nt) {
    const bf16* bp = b0 + (size_t)nt * 16 * K + k;
    t.b[nt].q[0] = *(const uint4*)(bp);
    t.b[nt].q[1] = *(const uint4*)(bp + 16);
  }
}

DEV void mma_tile(v8f acc[8], const Tile& t) {
  #pragma unroll
  for (int mt = 0; mt < 2; ++mt) {
    #pragma unroll
    for (int nt = 0; nt < 4; ++nt) {
      acc[mt * 4 + nt] = wmma_bf16(t.a[mt], t.b[nt], acc[mt * 4 + nt]);
    }
  }
}

__global__ __launch_bounds__(256) void gemm_bf16_wmma(const bf16* __restrict__ A,
                                                      const bf16* __restrict__ BT,
                                                      const float* __restrict__ bias,
                                                      void* __restrict__ out,
                                                      int M, int N, int K, int ldc,
                                                      int mode) {
  const int lane = threadIdx.x & 31;
  const int wave = threadIdx.x >> 5;
  const int half = lane >> 4;   // 0/1
  const int lm   = lane & 15;
  const int m0 = blockIdx.y * 128 + (wave >> 1) * 32;
  const int n0 = blockIdx.x * 128 + (wave & 1) * 64;

  const bf16* arow0 = A + (size_t)(m0 + lm) * K + half * 8;
  const bf16* arow1 = arow0 + (size_t)16 * K;
  const bf16* b0    = BT + (size_t)(n0 + lm) * K + half * 8;

  v8f acc[8];
  #pragma unroll
  for (int i = 0; i < 8; ++i) acc[i] = zero8();

  Tile t0, t1;
  load_tile(t0, arow0, arow1, b0, K, 0);
  // K is a multiple of 64 for every GEMM in this block (1024 or 4096)
  for (int k = 0; k < K; k += 64) {
    __builtin_prefetch(arow0 + k + 512, 0, 3);  // global_prefetch_b8
    load_tile(t1, arow0, arow1, b0, K, k + 32);  // in flight during t0 WMMAs
    mma_tile(acc, t0);
    if (k + 64 < K) load_tile(t0, arow0, arow1, b0, K, k + 64);
    mma_tile(acc, t1);
  }

  #pragma unroll
  for (int mt = 0; mt < 2; ++mt) {
    #pragma unroll
    for (int nt = 0; nt < 4; ++nt) {
      const int c = n0 + nt * 16 + lm;
      const float bv = bias ? bias[c] : 0.f;
      #pragma unroll
      for (int v = 0; v < 8; ++v) {
        const int r = m0 + mt * 16 + half * 8 + v;
        float val = acc[mt * 4 + nt][v] + bv;
        if (mode == 1) val = fmaxf(val, 0.f);
        if (mode == 2) {
          ((float*)out)[(size_t)r * ldc + c] = val;
        } else if (mode == 3) {
          int bb = r >> 11, ss = r & 2047, hh = c >> 6, ee = c & 63;
          ((bf16*)out)[(((size_t)((bb * Hh + hh) * DHd + ee)) << 11) + ss] = f2bf(val);
        } else {
          ((bf16*)out)[(size_t)r * ldc + c] = f2bf(val);
        }
      }
    }
  }
}

// ---------------------------------------------------------------------------
// Flash attention: Q,K bf16 [tok, D] (head h at cols h*64), VT bf16 [B,H,DH,S].
// One wave per 16-row query tile; online softmax; output concat f32 [tok, D].
// V fragments are loaded AFTER the K fragments but BEFORE the score WMMAs:
// in-order load completion means the score WMMAs only wait for K, while the
// V loads overlap the softmax VALU + LDS re-layout phase.
__global__ __launch_bounds__(256) void attn_wmma_kernel(const bf16* __restrict__ Q,
                                                        const bf16* __restrict__ Kt,
                                                        const bf16* __restrict__ VT,
                                                        float* __restrict__ Oc) {
  __shared__ bf16 lp[8][16 * 32];  // wave-private 16x32 bf16 P tile
  const int lane = threadIdx.x & 31;
  const int wave = threadIdx.x >> 5;
  const int half = lane >> 4;
  const int lm   = lane & 15;

  const int id = blockIdx.x * 8 + wave;      // 0..4095
  const int qt = id & 127;                   // query tile within (b,h)
  const int bh = id >> 7;
  const int b = bh >> 4, h = bh & 15;

  // Q fragments (16 x 64, two K-steps of 32)
  const bf16* qrow = Q + ((size_t)(b * Ss + qt * 16 + lm)) * Dm + h * DHd + half * 8;
  Frag qa[2];
  qa[0].q[0] = *(const uint4*)(qrow);
  qa[0].q[1] = *(const uint4*)(qrow + 16);
  qa[1].q[0] = *(const uint4*)(qrow + 32);
  qa[1].q[1] = *(const uint4*)(qrow + 48);

  const bf16* kbase = Kt + ((size_t)b * Ss) * Dm + h * DHd + half * 8;
  const bf16* vbase = VT + ((size_t)(b * Hh + h) * DHd) * Ss + half * 8;

  float mrow[8], lrow[8], ofac[8];
  v8f o[4];
  #pragma unroll
  for (int i = 0; i < 4; ++i) o[i] = zero8();
  #pragma unroll
  for (int v = 0; v < 8; ++v) { mrow[v] = -1e30f; lrow[v] = 0.f; }

  bf16* myp = lp[wave];

  for (int j = 0; j < Ss / 32; ++j) {  // 32 keys (two 16-key tiles) per step
    // ---- K fragments first ----
    Frag kf[4];  // [tile2][kk]
    #pragma unroll
    for (int t2 = 0; t2 < 2; ++t2) {
      #pragma unroll
      for (int kk = 0; kk < 2; ++kk) {
        const bf16* kp = kbase + (size_t)(j * 32 + t2 * 16 + lm) * Dm + kk * 32;
        kf[t2 * 2 + kk].q[0] = *(const uint4*)kp;
        kf[t2 * 2 + kk].q[1] = *(const uint4*)(kp + 16);
      }
    }
    // ---- V fragments issued now, consumed after softmax ----
    Frag vf[4];
    #pragma unroll
    for (int nt = 0; nt < 4; ++nt) {
      const bf16* vp = vbase + (size_t)(nt * 16 + lm) * Ss + j * 32;
      vf[nt].q[0] = *(const uint4*)vp;
      vf[nt].q[1] = *(const uint4*)(vp + 16);
    }

    v8f s0 = zero8(), s1 = zero8();
    s0 = wmma_bf16(qa[0], kf[0], s0);
    s0 = wmma_bf16(qa[1], kf[1], s0);
    s1 = wmma_bf16(qa[0], kf[2], s1);
    s1 = wmma_bf16(qa[1], kf[3], s1);

    // online softmax per row (rows live in 16-lane half-wave groups)
    #pragma unroll
    for (int v = 0; v < 8; ++v) {
      float a0 = s0[v] * 0.125f, a1 = s1[v] * 0.125f;  // 1/sqrt(64)
      float mx = fmaxf(a0, a1);
      #pragma unroll
      for (int off = 8; off; off >>= 1) mx = fmaxf(mx, __shfl_xor(mx, off, 32));
      float mn = fmaxf(mrow[v], mx);
      float p0 = __expf(a0 - mn), p1 = __expf(a1 - mn);
      float rs = p0 + p1;
      #pragma unroll
      for (int off = 8; off; off >>= 1) rs += __shfl_xor(rs, off, 32);
      float corr = __expf(mrow[v] - mn);
      lrow[v] = lrow[v] * corr + rs;
      mrow[v] = mn;
      ofac[v] = corr;
      myp[(half * 8 + v) * 32 + lm]      = f2bf(p0);   // C-layout -> A-layout via LDS
      myp[(half * 8 + v) * 32 + 16 + lm] = f2bf(p1);
    }
    #pragma unroll
    for (int nt = 0; nt < 4; ++nt) {
      #pragma unroll
      for (int v = 0; v < 8; ++v) o[nt][v] *= ofac[v];
    }

    asm volatile("s_wait_dscnt 0" ::: "memory");  // cross-lane P visible (in-wave DS order)

    Frag pa;
    const bf16* pr = myp + lm * 32 + half * 8;
    pa.q[0] = *(const uint4*)pr;
    pa.q[1] = *(const uint4*)(pr + 16);

    #pragma unroll
    for (int nt = 0; nt < 4; ++nt) o[nt] = wmma_bf16(pa, vf[nt], o[nt]);

    asm volatile("s_wait_dscnt 0" ::: "memory");  // WAR: P reads done before next store
  }

  #pragma unroll
  for (int nt = 0; nt < 4; ++nt) {
    #pragma unroll
    for (int v = 0; v < 8; ++v) {
      int tok = b * Ss + qt * 16 + half * 8 + v;
      int col = h * DHd + nt * 16 + lm;
      Oc[(size_t)tok * Dm + col] = o[nt][v] / lrow[v];
    }
  }
}

// ---------------------------------------------------------------------------
// LayerNorm over D=1024; optional residual add; optional f32/bf16 outputs.
__global__ __launch_bounds__(256) void ln_kernel(const float* __restrict__ a,
                                                 const float* __restrict__ b,
                                                 const float* __restrict__ g,
                                                 const float* __restrict__ be,
                                                 float* __restrict__ outf,
                                                 bf16* __restrict__ outb) {
  __shared__ float sh[256], sh2[256];
  const int row = blockIdx.x, t = threadIdx.x;
  const float* pa = a + (size_t)row * Dm;
  const float* pb = b ? b + (size_t)row * Dm : nullptr;
  float x4[4];
  float s = 0.f, sq = 0.f;
  #pragma unroll
  for (int i = 0; i < 4; ++i) {
    float xv = pa[t * 4 + i] + (pb ? pb[t * 4 + i] : 0.f);
    x4[i] = xv; s += xv; sq += xv * xv;
  }
  sh[t] = s; sh2[t] = sq; __syncthreads();
  for (int off = 128; off; off >>= 1) {
    if (t < off) { sh[t] += sh[t + off]; sh2[t] += sh2[t + off]; }
    __syncthreads();
  }
  float mean = sh[0] * (1.f / Dm);
  float var  = sh2[0] * (1.f / Dm) - mean * mean;
  float rstd = rsqrtf(var + 1e-5f);
  #pragma unroll
  for (int i = 0; i < 4; ++i) {
    int c = t * 4 + i;
    float y = (x4[i] - mean) * rstd * g[c] + be[c];
    if (outf) outf[(size_t)row * Dm + c] = y;
    if (outb) outb[(size_t)row * Dm + c] = f2bf(y);
  }
}

// ---------------------------------------------------------------------------
extern "C" void kernel_launch(void* const* d_in, const int* in_sizes, int n_in,
                              void* d_out, int out_size, void* d_ws, size_t ws_size,
                              hipStream_t stream) {
  (void)in_sizes; (void)n_in; (void)out_size; (void)ws_size;

  const float* x    = (const float*)d_in[0];
  const float* Wq   = (const float*)d_in[1];
  const float* bq   = (const float*)d_in[2];
  const float* Wk   = (const float*)d_in[3];
  const float* bk   = (const float*)d_in[4];
  const float* Wv   = (const float*)d_in[5];
  const float* bv   = (const float*)d_in[6];
  const float* Wo   = (const float*)d_in[7];
  const float* bo   = (const float*)d_in[8];
  const float* gcat = (const float*)d_in[9];
  const float* bcat = (const float*)d_in[10];
  const float* g1   = (const float*)d_in[11];
  const float* b1n  = (const float*)d_in[12];
  const float* W1   = (const float*)d_in[13];
  const float* b1   = (const float*)d_in[14];
  const float* W2   = (const float*)d_in[15];
  const float* b2   = (const float*)d_in[16];
  const float* g2   = (const float*)d_in[17];
  const float* b2n  = (const float*)d_in[18];

  const size_t MB = 1024ull * 1024ull;
  char* ws = (char*)d_ws;
  // ---- scratch arena (with lifetime-based reuse; peak 88 MB) ----
  bf16* WqT = (bf16*)(ws + 0 * MB);    // [1024,1024]
  bf16* WkT = (bf16*)(ws + 2 * MB);
  bf16* WvT = (bf16*)(ws + 4 * MB);
  bf16* WoT = (bf16*)(ws + 6 * MB);
  bf16* W1T = (bf16*)(ws + 8 * MB);    // [4096,1024]
  bf16* W2T = (bf16*)(ws + 16 * MB);   // [1024,4096]
  bf16* xb  = (bf16*)(ws + 24 * MB);   // [4096,1024]
  bf16* Qb  = (bf16*)(ws + 32 * MB);   // [4096,1024]
  bf16* Kb  = (bf16*)(ws + 40 * MB);
  bf16* VTb = (bf16*)(ws + 48 * MB);   // [B,H,DH,S]
  float* concatf  = (float*)(ws + 56 * MB);  // [4096,1024] f32
  bf16*  lncat    = (bf16*)(ws + 24 * MB);   // reuse (xb dead)
  float* attn_out = (float*)(ws + 32 * MB);  // reuse (Q,K dead)
  float* x1f      = (float*)(ws + 48 * MB);  // reuse (VT,concat dead)
  bf16*  x1b      = (bf16*)(ws + 64 * MB);
  bf16*  hb       = (bf16*)(ws + 24 * MB);   // reuse (lncat,attn_out dead) [4096,4096]
  float* y2       = (float*)(ws + 72 * MB);

  // 1) converts / weight transposes (one-time layout pass, ~4 us of HBM traffic)
  cvt_bf16_kernel<<<(NTOK * Dm) / 256, 256, 0, stream>>>(x, xb, NTOK * Dm);
  transpose_cvt_kernel<<<dim3((Dm * DHd) / 256, Hh), 256, 0, stream>>>(Wq, WqT, Dm, DHd);
  transpose_cvt_kernel<<<dim3((Dm * DHd) / 256, Hh), 256, 0, stream>>>(Wk, WkT, Dm, DHd);
  transpose_cvt_kernel<<<dim3((Dm * DHd) / 256, Hh), 256, 0, stream>>>(Wv, WvT, Dm, DHd);
  transpose_cvt_kernel<<<dim3((Dm * Dm) / 256, 1), 256, 0, stream>>>(Wo, WoT, Dm, Dm);
  transpose_cvt_kernel<<<dim3((Dm * DFFd) / 256, 1), 256, 0, stream>>>(W1, W1T, Dm, DFFd);
  transpose_cvt_kernel<<<dim3((Dm * DFFd) / 256, 1), 256, 0, stream>>>(W2, W2T, DFFd, Dm);

  // 2) QKV projections (Q/K in [tok,D] "concat" layout; V stored transposed)
  dim3 gD(Dm / 128, NTOK / 128);
  gemm_bf16_wmma<<<gD, 256, 0, stream>>>(xb, WqT, bq, Qb,  NTOK, Dm, Dm, Dm, 0);
  gemm_bf16_wmma<<<gD, 256, 0, stream>>>(xb, WkT, bk, Kb,  NTOK, Dm, Dm, Dm, 0);
  gemm_bf16_wmma<<<gD, 256, 0, stream>>>(xb, WvT, bv, VTb, NTOK, Dm, Dm, Dm, 3);

  // 3) attention (softmax(QK^T/8)V), writes concat f32
  attn_wmma_kernel<<<(NTOK / 16) / 8, 256, 0, stream>>>(Qb, Kb, VTb, concatf);

  // 4) LN on concat -> bf16
  ln_kernel<<<NTOK, 256, 0, stream>>>(concatf, nullptr, gcat, bcat, nullptr, lncat);

  // 5) output projection
  gemm_bf16_wmma<<<gD, 256, 0, stream>>>(lncat, WoT, bo, attn_out, NTOK, Dm, Dm, Dm, 2);

  // 6) x1 = LN(x + attn_out) -> f32 (residual) + bf16 (GEMM input)
  ln_kernel<<<NTOK, 256, 0, stream>>>(attn_out, x, g1, b1n, x1f, x1b);

  // 7) FFN
  dim3 gF(DFFd / 128, NTOK / 128);
  gemm_bf16_wmma<<<gF, 256, 0, stream>>>(x1b, W1T, b1, hb, NTOK, DFFd, Dm, DFFd, 1);
  gemm_bf16_wmma<<<gD, 256, 0, stream>>>(hb, W2T, b2, y2, NTOK, Dm, DFFd, Dm, 2);

  // 8) out = LN(x1 + ffn)
  ln_kernel<<<NTOK, 256, 0, stream>>>(y2, x1f, g2, b2n, (float*)d_out, nullptr);
}
